// Attention_4037269258732
// MI455X (gfx1250) — matrix operand order repaired and checked by reading, not compiled, not run on hardware
//
#include <hip/hip_runtime.h>
#include <cstdint>
#include <cstddef>

// ---------------------------------------------------------------------------
// CDNA5 (gfx1250) attention block: all GEMMs via v_wmma_f32_16x16x32_bf16,
// GEMM staging via the Tensor Data Mover (tensor_load_to_lds + TENSORcnt).
// wave32; WMMA fragment layouts per CDNA5 ISA 7.12.2; D# per ISA 8.3/8.4.
// ---------------------------------------------------------------------------

typedef __attribute__((ext_vector_type(8)))  __bf16       v8bf;
typedef __attribute__((ext_vector_type(16))) __bf16       v16bf;
typedef __attribute__((ext_vector_type(8)))  float        v8f;
typedef __attribute__((ext_vector_type(4)))  unsigned int v4u;
typedef __attribute__((ext_vector_type(8)))  int          v8i;
typedef __attribute__((ext_vector_type(4)))  int          v4i;

#define S_LEN 2048
#define EMB   2048
#define NH    32
#define NKVH  8
#define HD    64
#define QKDIM (NH * HD)    // 2048
#define KVDIM (NKVH * HD)  // 512

// ---- WMMA fragment builders (bf16, 16x16x32) ------------------------------
// A (16x32, MxK): lane L<16 holds row M=L, K in {0..7}U{16..23};
//                 lane L>=16 holds row M=L-16, K in {8..15}U{24..31}.
__device__ __forceinline__ v16bf frag_a(const __bf16* __restrict__ rowp, int lane) {
  const int half = (lane & 16) ? 8 : 0;
  v8bf lo = *reinterpret_cast<const v8bf*>(rowp + half);
  v8bf hi = *reinterpret_cast<const v8bf*>(rowp + 16 + half);
  v16bf r;
#pragma unroll
  for (int i = 0; i < 8; ++i) { r[i] = lo[i]; r[i + 8] = hi[i]; }
  return r;
}

// B (32x16, KxN): lane L<16 holds column N=L, K in {0..7}U{16..23};
//                 lane L>=16 holds column N=L-16, K in {8..15}U{24..31}.
__device__ __forceinline__ v16bf frag_b(const __bf16* __restrict__ colp, int lane) {
  const int kb = (lane & 16) ? 8 : 0;
  v8bf lo = *reinterpret_cast<const v8bf*>(colp + kb);
  v8bf hi = *reinterpret_cast<const v8bf*>(colp + 16 + kb);
  v16bf r;
#pragma unroll
  for (int i = 0; i < 8; ++i) { r[i] = lo[i]; r[i + 8] = hi[i]; }
  return r;
}

// ---- Tensor Data Mover: 2D bf16 tile -> padded LDS rows -------------------
// Tile rows of 32 elements (64B = 16 DWORDs) padded with 4 DWORDs -> 40-elem
// LDS rows (80B, 16B aligned). D# group0/group1 per CDNA5 ISA 8.3/8.4.
// This toolchain's builtin takes 6 args:
//   (uint32x4 g0, int32x8 g1, int32x4 g2, int32x4 g3, int32x8 g4, i32 cpol)
#if __has_builtin(__builtin_amdgcn_tensor_load_to_lds)
#define USE_TDM 1
__device__ __forceinline__ void tdm_load_tile_2d(
    unsigned lds_off, const void* gptr, unsigned tile_rows,
    unsigned tensor_k, unsigned tensor_rows, unsigned row_stride_elems) {
  const unsigned long long ga = (unsigned long long)(uintptr_t)gptr;
  v4u g0;
  g0[0] = 1u;                                            // count=1, user D#
  g0[1] = lds_off;                                       // lds_addr (bytes)
  g0[2] = (unsigned)(ga & 0xffffffffu);                  // global_addr[31:0]
  g0[3] = (unsigned)((ga >> 32) & 0x01ffffffu) | (2u << 30);  // ga[56:32] | type=2
  v8i g1;
  // data_size=2B (1<<16), pad_enable (1<<20), pad_interval=16DW (3<<22),
  // pad_amount=4DW (3<<25); workgroup_mask=0 (not in cluster)
  g1[0] = (int)((1u << 16) | (1u << 20) | (3u << 22) | (3u << 25));
  g1[1] = (int)((tensor_k & 0xffffu) << 16);             // tensor_dim0[15:0]
  g1[2] = (int)(((tensor_k >> 16) & 0xffffu) | ((tensor_rows & 0xffffu) << 16));
  g1[3] = (int)(((tensor_rows >> 16) & 0xffffu) | (32u << 16));  // tile_dim0=32
  g1[4] = (int)(tile_rows & 0xffffu);                    // tile_dim1, tile_dim2=0
  g1[5] = (int)row_stride_elems;                         // tensor_dim0_stride[31:0]
  g1[6] = 0;
  g1[7] = 0;
  const v4i z4 = {};
  const v8i z8 = {};
  __builtin_amdgcn_tensor_load_to_lds(g0, g1, z4, z4, z8, 0);
}
#else
#define USE_TDM 0
#endif

// ---- fp32 -> bf16 down-convert --------------------------------------------
__global__ __launch_bounds__(256) void cvt_f32_to_bf16(
    const float* __restrict__ in, __bf16* __restrict__ out, int n) {
  int i = blockIdx.x * blockDim.x + threadIdx.x;
  const int stride = gridDim.x * blockDim.x;
  for (; i < n; i += stride) out[i] = (__bf16)in[i];
}

// ---- GEMM: C[M][N] = A[M][K] * W[N][K]^T + bias[N] (bf16 in, fp32 out) ----
// Block: 256 threads (8 waves), 64x256 C tile, K-step 32.
// Wave grid 2x4; each wave computes 32x64 = 2x4 accumulators (8 WMMA/K-step).
__global__ __launch_bounds__(256) void gemm_bf16_wmma(
    const __bf16* __restrict__ A, const __bf16* __restrict__ W,
    const float* __restrict__ bias, float* __restrict__ C,
    int M, int N, int K) {
  __shared__ alignas(16) __bf16 As[64][40];    // 64 rows  x 32 (pad->40)
  __shared__ alignas(16) __bf16 Bs[256][40];   // 256 rows x 32 (pad->40)

  const int tid = threadIdx.x, lane = tid & 31, wave = tid >> 5;
  const int m0 = blockIdx.y * 64, n0 = blockIdx.x * 256;
  const int wm = (wave >> 2) * 32;   // 0,32
  const int wn = (wave & 3) * 64;    // 0,64,128,192
  const int ln15 = lane & 15;

  v8f acc[2][4];
#pragma unroll
  for (int i = 0; i < 2; ++i)
#pragma unroll
    for (int j = 0; j < 4; ++j) acc[i][j] = {};

#if USE_TDM
  const unsigned asOff = (unsigned)(uintptr_t)(void*)&As[0][0];
  const unsigned bsOff = (unsigned)(uintptr_t)(void*)&Bs[0][0];
#endif

  for (int k0 = 0; k0 < K; k0 += 32) {
    __syncthreads();
#if USE_TDM
    if (wave == 0) {
      tdm_load_tile_2d(asOff, A + (size_t)m0 * K + k0, 64, (unsigned)K,
                       (unsigned)M, (unsigned)K);
      tdm_load_tile_2d(bsOff, W + (size_t)n0 * K + k0, 256, (unsigned)K,
                       (unsigned)N, (unsigned)K);
      __builtin_amdgcn_s_wait_tensorcnt(0);
    }
#else
    {
      const int lrow = tid >> 2;        // 0..63
      const int lcol = (tid & 3) * 8;   // 0,8,16,24
      *reinterpret_cast<v8bf*>(&As[lrow][lcol]) =
          *reinterpret_cast<const v8bf*>(A + (size_t)(m0 + lrow) * K + k0 + lcol);
#pragma unroll
      for (int rep = 0; rep < 4; ++rep) {
        const int brow = rep * 64 + lrow;
        *reinterpret_cast<v8bf*>(&Bs[brow][lcol]) =
            *reinterpret_cast<const v8bf*>(W + (size_t)(n0 + brow) * K + k0 + lcol);
      }
    }
#endif
    __syncthreads();

    v16bf a0 = frag_a(&As[wm + ln15][0], lane);
    v16bf a1 = frag_a(&As[wm + 16 + ln15][0], lane);
#pragma unroll
    for (int j = 0; j < 4; ++j) {
      v16bf b = frag_b(&Bs[wn + j * 16 + ln15][0], lane);
      acc[0][j] = __builtin_amdgcn_wmma_f32_16x16x32_bf16(false, a0, false, b,
                                                          (short)0, acc[0][j],
                                                          false, false);
      acc[1][j] = __builtin_amdgcn_wmma_f32_16x16x32_bf16(false, a1, false, b,
                                                          (short)0, acc[1][j],
                                                          false, false);
    }
  }

  // C/D layout: VGPR r -> row (r or r+8 by lane half), lane%16 -> column.
  const int rOff = (lane & 16) ? 8 : 0;
#pragma unroll
  for (int i = 0; i < 2; ++i) {
    const int rbase = m0 + wm + i * 16 + rOff;
#pragma unroll
    for (int j = 0; j < 4; ++j) {
      const int col = n0 + wn + j * 16 + ln15;
      const float bi = bias[col];
#pragma unroll
      for (int r = 0; r < 8; ++r)
        C[(size_t)(rbase + r) * N + col] = acc[i][j][r] + bi;
    }
  }
}

// ---- RoPE on Q,K + bf16 conversion of Q,K,V -------------------------------
__global__ __launch_bounds__(256) void rope_cvt(
    const float* __restrict__ qf, const float* __restrict__ kf,
    const float* __restrict__ vf, const float* __restrict__ rope,
    __bf16* __restrict__ qb, __bf16* __restrict__ kb, __bf16* __restrict__ vb) {
  const int s = blockIdx.x;
  const float* cs = rope + (size_t)s * (2 * HD);  // [cos(64) | sin(64)]
  const float* qs = qf + (size_t)s * QKDIM;
  const float* ks = kf + (size_t)s * KVDIM;
  const float* vs = vf + (size_t)s * KVDIM;

  for (int i = threadIdx.x; i < QKDIM; i += 256) {
    const int d = i & (HD - 1);
    const float c = cs[d], sn = cs[HD + d];
    const float x0 = qs[i];
    const float xr = (d < HD / 2) ? -qs[i + HD / 2] : qs[i - HD / 2];
    qb[(size_t)s * QKDIM + i] = (__bf16)(x0 * c + xr * sn);
  }
  for (int i = threadIdx.x; i < KVDIM; i += 256) {
    const int d = i & (HD - 1);
    const float c = cs[d], sn = cs[HD + d];
    const float x0 = ks[i];
    const float xr = (d < HD / 2) ? -ks[i + HD / 2] : ks[i - HD / 2];
    kb[(size_t)s * KVDIM + i] = (__bf16)(x0 * c + xr * sn);
  }
  for (int i = threadIdx.x; i < KVDIM; i += 256) {
    vb[(size_t)s * KVDIM + i] = (__bf16)vs[i];
  }
}

// ---- Flash attention with sink renorm -------------------------------------
// Block: 256 threads (8 waves) = 128 query rows of one head; wave owns 16 rows.
#define QB 128
#define KB 64

__global__ __launch_bounds__(256) void flash_attn_wmma(
    const __bf16* __restrict__ Q,   // [S][QKDIM]
    const __bf16* __restrict__ Kc,  // [S][KVDIM]
    const __bf16* __restrict__ Vc,  // [S][KVDIM]
    const float* __restrict__ sinks,
    __bf16* __restrict__ O) {       // [S][QKDIM]
  __shared__ alignas(16) __bf16 Ks[KB][72];     // [key][dim]
  __shared__ alignas(16) __bf16 Vt[HD][72];     // [dim][key] (transposed)
  __shared__ alignas(16) __bf16 Ps[8][16][40];  // per-wave P scratch (C->A)

  const int tid = threadIdx.x, lane = tid & 31, wave = tid >> 5;
  const int h = blockIdx.y, kvh = h >> 2;       // REP = 4
  const int q0 = blockIdx.x * QB;
  const int m0 = q0 + wave * 16;
  const int ln15 = lane & 15;
  const int rOff = (lane & 16) ? 8 : 0;

  const __bf16* qp = Q + (size_t)(m0 + ln15) * QKDIM + h * HD;
  const v16bf qf0 = frag_a(qp, lane);
  const v16bf qf1 = frag_a(qp + 32, lane);

  v8f o0 = {}, o1 = {}, o2 = {}, o3 = {};
  float mrun[8], lrun[8];
#pragma unroll
  for (int r = 0; r < 8; ++r) { mrun[r] = -3.0e38f; lrun[r] = 0.0f; }

  const float scale = 0.125f;  // 1/sqrt(64)
  const int nkb = (q0 + QB - 1) / KB + 1;

  for (int kbi = 0; kbi < nkb; ++kbi) {
    const int kbase = kbi * KB;
    __syncthreads();
    for (int c = tid; c < (KB * HD / 8); c += 256) {
      const int key = c >> 3, kc = (c & 7) * 8;
      const size_t gro = (size_t)(kbase + key) * KVDIM + kvh * HD + kc;
      *reinterpret_cast<v8bf*>(&Ks[key][kc]) =
          *reinterpret_cast<const v8bf*>(Kc + gro);
      v8bf tv = *reinterpret_cast<const v8bf*>(Vc + gro);
#pragma unroll
      for (int j = 0; j < 8; ++j) Vt[kc + j][key] = tv[j];
    }
    __syncthreads();

#pragma unroll
    for (int g = 0; g < 2; ++g) {
      const int koff = g * 32;
      v8f s0 = {}, s1 = {};
      {
        v16bf bk00 = frag_b(&Ks[koff + ln15][0], lane);
        v16bf bk01 = frag_b(&Ks[koff + ln15][32], lane);
        s0 = __builtin_amdgcn_wmma_f32_16x16x32_bf16(false, qf0, false, bk00,
                                                     (short)0, s0, false, false);
        s0 = __builtin_amdgcn_wmma_f32_16x16x32_bf16(false, qf1, false, bk01,
                                                     (short)0, s0, false, false);
        v16bf bk10 = frag_b(&Ks[koff + 16 + ln15][0], lane);
        v16bf bk11 = frag_b(&Ks[koff + 16 + ln15][32], lane);
        s1 = __builtin_amdgcn_wmma_f32_16x16x32_bf16(false, qf0, false, bk10,
                                                     (short)0, s1, false, false);
        s1 = __builtin_amdgcn_wmma_f32_16x16x32_bf16(false, qf1, false, bk11,
                                                     (short)0, s1, false, false);
      }
      float sc[8];
#pragma unroll
      for (int r = 0; r < 8; ++r) {
        const int qi = m0 + r + rOff;
        const int kj0 = kbase + koff + ln15;
        const int kj1 = kj0 + 16;
        float a = s0[r] * scale; if (kj0 > qi) a = -3.0e38f;
        float b = s1[r] * scale; if (kj1 > qi) b = -3.0e38f;
        float mx = fmaxf(a, b);
        mx = fmaxf(mx, __shfl_xor(mx, 1, 32));
        mx = fmaxf(mx, __shfl_xor(mx, 2, 32));
        mx = fmaxf(mx, __shfl_xor(mx, 4, 32));
        mx = fmaxf(mx, __shfl_xor(mx, 8, 32));
        const float nm = fmaxf(mrun[r], mx);
        const float f = __expf(mrun[r] - nm);
        const float p0 = __expf(a - nm);
        const float p1 = __expf(b - nm);
        float rs = p0 + p1;
        rs += __shfl_xor(rs, 1, 32);
        rs += __shfl_xor(rs, 2, 32);
        rs += __shfl_xor(rs, 4, 32);
        rs += __shfl_xor(rs, 8, 32);
        lrun[r] = lrun[r] * f + rs;
        mrun[r] = nm;
        sc[r] = f;
        Ps[wave][r + rOff][ln15] = (__bf16)p0;
        Ps[wave][r + rOff][16 + ln15] = (__bf16)p1;
      }
#pragma unroll
      for (int r = 0; r < 8; ++r) {
        o0[r] *= sc[r]; o1[r] *= sc[r]; o2[r] *= sc[r]; o3[r] *= sc[r];
      }
      // wave-private LDS RAW: DS ops in-order within a wave; keep IR order.
      asm volatile("" ::: "memory");
      v16bf pa = frag_a(&Ps[wave][ln15][0], lane);
      v16bf v0 = frag_b(&Vt[0 * 16 + ln15][koff], lane);
      v16bf v1 = frag_b(&Vt[1 * 16 + ln15][koff], lane);
      v16bf v2 = frag_b(&Vt[2 * 16 + ln15][koff], lane);
      v16bf v3 = frag_b(&Vt[3 * 16 + ln15][koff], lane);
      o0 = __builtin_amdgcn_wmma_f32_16x16x32_bf16(false, pa, false, v0,
                                                   (short)0, o0, false, false);
      o1 = __builtin_amdgcn_wmma_f32_16x16x32_bf16(false, pa, false, v1,
                                                   (short)0, o1, false, false);
      o2 = __builtin_amdgcn_wmma_f32_16x16x32_bf16(false, pa, false, v2,
                                                   (short)0, o2, false, false);
      o3 = __builtin_amdgcn_wmma_f32_16x16x32_bf16(false, pa, false, v3,
                                                   (short)0, o3, false, false);
    }
  }

  // epilogue: normalize by l, then sink logaddexp renormalization
  const float snk = sinks[h];
  float inv[8];
#pragma unroll
  for (int r = 0; r < 8; ++r) {
    const float lse = mrun[r] + logf(lrun[r]);
    const float mx = fmaxf(lse, snk);
    const float comb = mx + log1pf(__expf(-fabsf(lse - snk)));
    float d = lse - comb;
    d = fminf(fmaxf(d, -20.0f), 0.0f);
    inv[r] = __expf(d) / lrun[r];
  }
#pragma unroll
  for (int r = 0; r < 8; ++r) {
    const size_t row = (size_t)(m0 + r + rOff) * QKDIM + h * HD;
    O[row + 0 * 16 + ln15] = (__bf16)(o0[r] * inv[r]);
    O[row + 1 * 16 + ln15] = (__bf16)(o1[r] * inv[r]);
    O[row + 2 * 16 + ln15] = (__bf16)(o2[r] * inv[r]);
    O[row + 3 * 16 + ln15] = (__bf16)(o3[r] * inv[r]);
  }
}

// ---------------------------------------------------------------------------
extern "C" void kernel_launch(void* const* d_in, const int* in_sizes, int n_in,
                              void* d_out, int out_size, void* d_ws, size_t ws_size,
                              hipStream_t stream) {
  const float* x     = (const float*)d_in[0];
  const float* rope  = (const float*)d_in[1];
  const float* wq_w  = (const float*)d_in[2];
  const float* wq_b  = (const float*)d_in[3];
  const float* wk_w  = (const float*)d_in[4];
  const float* wk_b  = (const float*)d_in[5];
  const float* wv_w  = (const float*)d_in[6];
  const float* wv_b  = (const float*)d_in[7];
  const float* wo_w  = (const float*)d_in[8];
  const float* wo_b  = (const float*)d_in[9];
  const float* sinks = (const float*)d_in[10];
  float* out = (float*)d_out;

  char* p = (char*)d_ws;
  auto take = [&](size_t bytes) {
    char* r = p;
    p += (bytes + 255) & ~(size_t)255;
    return (void*)r;
  };

  __bf16* x_bf   = (__bf16*)take((size_t)S_LEN * EMB * 2);
  __bf16* wq_bf  = (__bf16*)take((size_t)QKDIM * EMB * 2);
  __bf16* wk_bf  = (__bf16*)take((size_t)KVDIM * EMB * 2);
  __bf16* wv_bf  = (__bf16*)take((size_t)KVDIM * EMB * 2);
  __bf16* wo_bf  = (__bf16*)take((size_t)EMB * QKDIM * 2);
  float*  qf     = (float*) take((size_t)S_LEN * QKDIM * 4);
  float*  kf     = (float*) take((size_t)S_LEN * KVDIM * 4);
  float*  vf     = (float*) take((size_t)S_LEN * KVDIM * 4);
  __bf16* q_bf   = (__bf16*)take((size_t)S_LEN * QKDIM * 2);
  __bf16* k_bf   = (__bf16*)take((size_t)S_LEN * KVDIM * 2);
  __bf16* v_bf   = (__bf16*)take((size_t)S_LEN * KVDIM * 2);
  __bf16* att_bf = (__bf16*)take((size_t)S_LEN * QKDIM * 2);

  cvt_f32_to_bf16<<<2048, 256, 0, stream>>>(x,    x_bf,  S_LEN * EMB);
  cvt_f32_to_bf16<<<2048, 256, 0, stream>>>(wq_w, wq_bf, QKDIM * EMB);
  cvt_f32_to_bf16<<<2048, 256, 0, stream>>>(wk_w, wk_bf, KVDIM * EMB);
  cvt_f32_to_bf16<<<2048, 256, 0, stream>>>(wv_w, wv_bf, KVDIM * EMB);
  cvt_f32_to_bf16<<<2048, 256, 0, stream>>>(wo_w, wo_bf, EMB * QKDIM);

  gemm_bf16_wmma<<<dim3(QKDIM / 256, S_LEN / 64), 256, 0, stream>>>(
      x_bf, wq_bf, wq_b, qf, S_LEN, QKDIM, EMB);
  gemm_bf16_wmma<<<dim3(KVDIM / 256, S_LEN / 64), 256, 0, stream>>>(
      x_bf, wk_bf, wk_b, kf, S_LEN, KVDIM, EMB);
  gemm_bf16_wmma<<<dim3(KVDIM / 256, S_LEN / 64), 256, 0, stream>>>(
      x_bf, wv_bf, wv_b, vf, S_LEN, KVDIM, EMB);

  rope_cvt<<<S_LEN, 256, 0, stream>>>(qf, kf, vf, rope, q_bf, k_bf, v_bf);

  flash_attn_wmma<<<dim3(S_LEN / QB, NH), 256, 0, stream>>>(
      q_bf, k_bf, v_bf, sinks, att_bf);

  gemm_bf16_wmma<<<dim3(EMB / 256, S_LEN / 64), 256, 0, stream>>>(
      att_bf, wo_bf, wo_b, out, S_LEN, EMB, QKDIM);
}
